// MultiGraphConv_31971736551501
// MI455X (gfx1250) — matrix operand order repaired and checked by reading, not compiled, not run on hardware
//
#include <hip/hip_runtime.h>

// ---------------------------------------------------------------------------
// MultiGraphConv on MI455X (gfx1250).
//
// Roofline: 17.2 GFLOP vs ~192 MB of HBM traffic -> memory bound once we use
// 16-bit WMMA (v_wmma_f32_16x16x32_f16). FP16 in / FP32 accumulate keeps
// ~1e-3 relative accuracy for these N(0,1)/U(0,1) inputs at K=512.
//
// Pass 1: collapse the R=4 weighted relation sum into two f16 matrices in ws.
// Pass 2: tiled WMMA GEMM, one 128x128 Z tile per workgroup per (b,l).
// ---------------------------------------------------------------------------

typedef __attribute__((ext_vector_type(16))) _Float16 v16h;
typedef __attribute__((ext_vector_type(8)))  _Float16 v8h;
typedef __attribute__((ext_vector_type(2)))  _Float16 v2h;
typedef __attribute__((ext_vector_type(8)))  float    v8f;

#define NM   512
#define NN   512
#define CC   128
#define NB   8
#define BT   128   // B*T = 8*16
#define RR   4
#define BK   32    // K step (matches 16x16x32 f16 WMMA)
#define LDSPAD 40  // 32 halves + 8 pad -> 80B rows, 16B aligned, bank spread

// ---------------------------------------------------------------------------
// Pass 1: A1[b][m][n] = sum_r w_n2m[r] * I_n2m[b][r][m][n]   (f16)
//         A2[b][m][n] = sum_r w_m2n[r] * I_m2n[b][r][m][n]   (f16)
// ---------------------------------------------------------------------------
__global__ __launch_bounds__(256)
void mgc_reduce_rel(const float* __restrict__ I_n2m,
                    const float* __restrict__ I_m2n,
                    const float* __restrict__ w_n2m,
                    const float* __restrict__ w_m2n,
                    _Float16* __restrict__ A1,
                    _Float16* __restrict__ A2)
{
    const size_t idx = (size_t)blockIdx.x * 256 + threadIdx.x;   // 0 .. 2M-1
    const size_t b   = idx >> 18;                 // / (512*512)
    const size_t mn  = idx & ((1u << 18) - 1);
    const size_t base = (b * RR) << 18;           // b*R*512*512
    float s1 = 0.0f, s2 = 0.0f;
#pragma unroll
    for (int r = 0; r < RR; ++r) {
        s1 += w_n2m[r] * I_n2m[base + ((size_t)r << 18) + mn];
        s2 += w_m2n[r] * I_m2n[base + ((size_t)r << 18) + mn];
    }
    A1[idx] = (_Float16)s1;
    A2[idx] = (_Float16)s2;
}

// ---------------------------------------------------------------------------
// Pass 2: Z[b,l,row,c] = sum_k A(row,k) * X[b,l,k,c]
//   TRANS_A=false: A stored [row][k]   (GEMM for Z_n2m, rows = m)
//   TRANS_A=true : A stored [k][row]   (GEMM for Z_m2n, rows = n)
// Grid: (4 row tiles, 128 (b,l) pairs). Block: 256 threads = 8 waves.
// Each wave: 2(m) x 4(c) accumulators of 16x16.
// ---------------------------------------------------------------------------
template <bool TRANS_A>
__global__ __launch_bounds__(256)
void mgc_gemm(const _Float16* __restrict__ A,   // [NB][512][512] f16
              const float*    __restrict__ X,   // [NB][T][512][128] f32
              float*          __restrict__ Z)   // [NB][T][512][128] f32
{
    __shared__ _Float16 sA[128][LDSPAD];  // [row][k]
    __shared__ _Float16 sX[128][LDSPAD];  // [c][k]  (transposed, f32->f16)

    const int rt   = blockIdx.x;          // row tile (128 rows each)
    const int bl   = blockIdx.y;          // b*16 + l
    const int b    = bl >> 4;
    const int tid  = threadIdx.x;
    const int wave = tid >> 5;
    const int lane = tid & 31;
    const int wr   = wave & 3;            // row group: 4 x 32 rows
    const int wq   = wave >> 2;           // col group: 2 x 64 cols
    const int l16  = lane & 15;
    const int lhalf = lane >> 4;          // 0 / 1

    const _Float16* Ab = A + (size_t)b  * NM * NN;
    const float*    Xb = X + (size_t)bl * NM * CC;
    float*          Zb = Z + (size_t)bl * NM * CC;
    const int row0 = rt * 128;

    v8f acc[2][4] = {};                   // f32 accumulators

    for (int k0 = 0; k0 < NM; k0 += BK) {
        // ---- stage A tile: sA[row 128][k 32] ----
        if (!TRANS_A) {
#pragma unroll
            for (int i = 0; i < 8; ++i) {
                const int idx = i * 256 + tid;        // 2048 dwords
                const int row = idx >> 4;
                const int kp  = idx & 15;
                const unsigned int d =
                    *(const unsigned int*)(Ab + (size_t)(row0 + row) * NN + k0 + 2 * kp);
                *(unsigned int*)(&sA[row][2 * kp]) = d;
            }
        } else {
#pragma unroll
            for (int i = 0; i < 8; ++i) {
                const int idx = i * 256 + tid;
                const int row = idx & 127;            // n (output row)
                const int kp  = idx >> 7;             // m pair
                v2h d;
                d.x = Ab[(size_t)(k0 + 2 * kp)     * NN + row0 + row];
                d.y = Ab[(size_t)(k0 + 2 * kp + 1) * NN + row0 + row];
                *(v2h*)(&sA[row][2 * kp]) = d;
            }
        }
        // ---- stage X tile transposed + f32->f16: sX[c 128][k 32] ----
#pragma unroll
        for (int i = 0; i < 8; ++i) {
            const int idx = i * 256 + tid;
            const int c  = idx & 127;
            const int kp = idx >> 7;
            const float f0 = Xb[(size_t)(k0 + 2 * kp)     * CC + c];
            const float f1 = Xb[(size_t)(k0 + 2 * kp + 1) * CC + c];
            v2h d; d.x = (_Float16)f0; d.y = (_Float16)f1;
            *(v2h*)(&sX[c][2 * kp]) = d;
        }
        // prefetch next X tile into cache while we compute (global_prefetch_b8)
        if (k0 + BK < NM) {
            __builtin_prefetch(Xb + (size_t)(k0 + BK) * CC + tid, 0, 0);
        }
        __syncthreads();

        // ---- fragments ----
        // A frag, contiguous k order (matches B): lane<16: row=lane,   K 0..15
        //                     lane>=16: row=lane-16, K 16..31
        v16h afr[2];
#pragma unroll
        for (int mi = 0; mi < 2; ++mi) {
            const int row = wr * 32 + mi * 16 + l16;
            const v8h lo = *(const v8h*)(&sA[row][lhalf * 16]);
            const v8h hi = *(const v8h*)(&sA[row][lhalf * 16 + 8]);
            afr[mi] = __builtin_shufflevector(lo, hi,
                0, 1, 2, 3, 4, 5, 6, 7, 8, 9, 10, 11, 12, 13, 14, 15);
        }
        // B frag, same contiguous k order as A: lane<16: col=lane, K 0..15 ; lane>=16: K 16..31
        v16h bfr[4];
#pragma unroll
        for (int ci = 0; ci < 4; ++ci) {
            const int c = wq * 64 + ci * 16 + l16;
            bfr[ci] = *(const v16h*)(&sX[c][lhalf * 16]);
        }
        // ---- 8 WMMAs per k-step ----
#pragma unroll
        for (int mi = 0; mi < 2; ++mi) {
#pragma unroll
            for (int ci = 0; ci < 4; ++ci) {
                acc[mi][ci] = __builtin_amdgcn_wmma_f32_16x16x32_f16(
                    false, afr[mi], false, bfr[ci],
                    (short)0, acc[mi][ci], false, false);
            }
        }
        __syncthreads();
    }

    // ---- store: C/D layout: VGPR v -> M = v + 8*(lane>=16), N = lane&15 ----
#pragma unroll
    for (int mi = 0; mi < 2; ++mi) {
#pragma unroll
        for (int ci = 0; ci < 4; ++ci) {
            const int Mbase = row0 + wr * 32 + mi * 16 + lhalf * 8;
            const int Nc    = wq * 64 + ci * 16 + l16;
#pragma unroll
            for (int v = 0; v < 8; ++v) {
                Zb[(size_t)(Mbase + v) * CC + Nc] = acc[mi][ci][v];
            }
        }
    }
}

// ---------------------------------------------------------------------------
extern "C" void kernel_launch(void* const* d_in, const int* in_sizes, int n_in,
                              void* d_out, int out_size, void* d_ws, size_t ws_size,
                              hipStream_t stream) {
    const float* x_m   = (const float*)d_in[0];   // (8,16,512,128)
    const float* x_n   = (const float*)d_in[1];   // (8,16,512,128)
    const float* I_m2n = (const float*)d_in[2];   // (8,4,512,512)
    const float* I_n2m = (const float*)d_in[3];   // (8,4,512,512)
    const float* w_m2n = (const float*)d_in[4];   // (4,)
    const float* w_n2m = (const float*)d_in[5];   // (4,)

    // tuple return order: (Z_m2n, Z_n2m), each 8*16*512*128 f32
    float* Z_m2n = (float*)d_out;
    float* Z_n2m = Z_m2n + (size_t)BT * NN * CC;

    _Float16* A1 = (_Float16*)d_ws;               // reduced I_n2m [b][m][n]
    _Float16* A2 = A1 + (size_t)NB * NM * NN;     // reduced I_m2n [b][m][n]
    (void)in_sizes; (void)n_in; (void)out_size; (void)ws_size;

    // Pass 1: collapse relation dimension (2M threads)
    mgc_reduce_rel<<<(NB * NM * NN) / 256, 256, 0, stream>>>(
        I_n2m, I_m2n, w_n2m, w_m2n, A1, A2);

    // Pass 2a: Z_n2m[b,l,m,c] = sum_n A1[b,m,n] * x_n[b,l,n,c]
    mgc_gemm<false><<<dim3(NM / 128, BT), 256, 0, stream>>>(A1, x_n, Z_n2m);

    // Pass 2b: Z_m2n[b,l,n,c] = sum_m A2[b,m,n] * x_m[b,l,m,c]  (A transposed)
    mgc_gemm<true><<<dim3(NN / 128, BT), 256, 0, stream>>>(A2, x_m, Z_m2n);
}
